// state_space_86629490360947
// MI455X (gfx1250) — matrix operand order repaired and checked by reading, not compiled, not run on hardware
//
#include <hip/hip_runtime.h>
#include <hip/hip_bf16.h>

typedef __attribute__((ext_vector_type(16))) _Float16 v16h;
typedef __attribute__((ext_vector_type(8)))  float    v8f;

#define B_    1024
#define T_    256
#define FULL_ 64
#define H_    256
#define A_    4
#define C_    32
#define ZD_   5
#define HID_  128
#define NL_   8

// Launder a uniform value through an empty asm so the compiler cannot prove
// weight-tile loads are loop-invariant (prevents LICM hoisting ~800 VGPRs of
// weights out of the time loop -> scratch spills). Runtime cost: zero.
#define LAUNDER(x) asm volatile("" : "+s"(x))

// ---------------------------------------------------------------------------
// A-fragment LDS addressing for V_WMMA_F32_16X16X32_F16 (wave32).
//   lanes 0-15 : M=lane,    halves 0-7 -> K 0..7,   halves 8-15 -> K 16..23
//   lanes 16-31: M=lane-16, halves 0-7 -> K 8..15,  halves 8-15 -> K 24..31
__device__ __forceinline__ int a_addr(int m, int k) {
  return ((m + ((k & 8) ? 16 : 0)) << 4) + ((k & 7) | ((k & 16) >> 1));
}

// B-tile layout: tile = 512 halves; lane l holds column N=l%16,
// K = (q/8)*16 + (l/16)*8 + q%8. Each lane loads 16 contiguous halves (32B).

__device__ __forceinline__ v8f wmma16(v16h a, v16h b, v8f c) {
  return __builtin_amdgcn_wmma_f32_16x16x32_f16(false, a, false, b, (short)0, c,
                                                false, false);
}

__device__ __forceinline__ float sigmoidf_(float x) { return 1.f / (1.f + __expf(-x)); }
__device__ __forceinline__ float eluf_(float x)     { return x > 0.f ? x : (__expf(x) - 1.f); }

// ---------------------------------------------------------------------------
// Weight conversion kernel: f32 row-major -> f16 WMMA B-tiles in ws.
//   [  0,160) WrB : combined r-gate K=320 ([Wih 0..255 | Whh 0..255]), tile=kt*16+nt
//   [160,320) WzB : combined z-gate (rows 256..511)
//   [320,352) WniB: n-gate x-part (Wih rows 512..767, K=64)
//   [352,480) WnhB: n-gate h-part (Whh rows 512..767, K=256)
//   [480,560) WinB: t_win K=320 (297 real, rest 0), tile=kt*8+nt
//   [560,816) WhB : t_wh, tile=L*32+kt*8+nt
__global__ void cvt_weights(const float* __restrict__ gru_whh,
                            const float* __restrict__ gru_wih,
                            const float* __restrict__ t_wh,
                            const float* __restrict__ t_win,
                            _Float16* __restrict__ w) {
  int i = blockIdx.x * 256 + threadIdx.x;
  if (i >= 816 * 512) return;
  int lane = (i >> 4) & 31;
  int q    = i & 15;
  int tile = i >> 9;
  int kin  = ((q & 8) << 1) + ((lane >> 4) << 3) + (q & 7);
  int ncol = lane & 15;
  float v = 0.f;
  if (tile < 320) {
    int gate = tile / 160;
    int tt = tile % 160;
    int kt = tt >> 4, nt = tt & 15;
    int n = gate * 256 + nt * 16 + ncol;
    int K = kt * 32 + kin;
    v = (K < 64) ? gru_wih[n * 64 + K] : gru_whh[n * 256 + (K - 64)];
  } else if (tile < 352) {
    int tt = tile - 320;
    int kt = tt >> 4, nt = tt & 15;
    int n = 512 + nt * 16 + ncol;
    int K = kt * 32 + kin;
    v = gru_wih[n * 64 + K];
  } else if (tile < 480) {
    int tt = tile - 352;
    int kt = tt >> 4, nt = tt & 15;
    int n = 512 + nt * 16 + ncol;
    int K = kt * 32 + kin;
    v = gru_whh[n * 256 + K];
  } else if (tile < 560) {
    int tt = tile - 480;
    int kt = tt >> 3, nt = tt & 7;
    int n = nt * 16 + ncol;
    int K = kt * 32 + kin;
    v = (K < 297) ? t_win[n * 297 + K] : 0.f;
  } else {
    int tt = tile - 560;
    int L = tt >> 5, r = tt & 31;
    int kt = r >> 3, nt = r & 7;
    int n = nt * 16 + ncol;
    int K = kt * 32 + kin;
    v = t_wh[(L * 128 + n) * 128 + K];
  }
  w[i] = (_Float16)v;
}

// ---------------------------------------------------------------------------
// Fused state-space kernel: one workgroup = 16 batch rows, 8 waves,
// sequential over T; all recurrent state lives in LDS; weights stream from L2.
__global__ __launch_bounds__(256) void ssm_main(
    const float* __restrict__ traj, const float* __restrict__ treat,
    const float* __restrict__ consts, const int* __restrict__ seq_lens,
    const float* __restrict__ gru_bih, const float* __restrict__ gru_bhh,
    const float* __restrict__ pe_w0, const float* __restrict__ pe_b0,
    const float* __restrict__ pe_wh, const float* __restrict__ pe_bh,
    const float* __restrict__ pe_wh0, const float* __restrict__ pe_bh0,
    const float* __restrict__ pe_wz0, const float* __restrict__ pe_bz0,
    const float* __restrict__ t_bin, const float* __restrict__ t_bh,
    const float* __restrict__ t_wz, const float* __restrict__ t_bz,
    const _Float16* __restrict__ W, float* __restrict__ out) {
  __shared__ __align__(32) _Float16 hA[8 * 512];
  __shared__ __align__(32) _Float16 xA[2 * 512];
  __shared__ __align__(32) _Float16 inA[10 * 512];
  __shared__ __align__(32) _Float16 actA[4 * 512];
  __shared__ __align__(32) _Float16 actB[4 * 512];
  __shared__ float h32[16 * 256];
  __shared__ float act32[16 * 128];
  __shared__ float enc0[16 * 64];
  __shared__ float enc1[16 * 64];
  __shared__ float cons32[16 * 32];
  __shared__ float zbuf[16 * 5];
  __shared__ float aprev[16 * 4];
  __shared__ int   slen[16];

  const int tid  = threadIdx.x;
  const int lane = tid & 31;
  const int wv   = tid >> 5;
  const int b0   = blockIdx.x * 16;
  const v16h* Wv = (const v16h*)W;

  // ---- patient encoder (f32 scalar, tiny) ----
  for (int e = tid; e < 16 * 32; e += 256) {
    int m = e >> 5, c = e & 31;
    cons32[e] = consts[(b0 + m) * C_ + c];
  }
  if (tid < 16) slen[tid] = seq_lens[b0 + tid];
  if (tid < 64) aprev[tid] = 0.f;
  __syncthreads();
  for (int e = tid; e < 1024; e += 256) {
    int m = e >> 6, h = e & 63;
    float s = pe_b0[h];
    for (int c = 0; c < 32; ++c) s += cons32[m * 32 + c] * pe_w0[h * 32 + c];
    enc0[e] = eluf_(s);
  }
  __syncthreads();
  float* src = enc0;
  float* dst = enc1;
  for (int i = 0; i < 3; ++i) {
    for (int e = tid; e < 1024; e += 256) {
      int m = e >> 6, h = e & 63;
      float s = pe_bh[i * 64 + h];
      const float* wr = &pe_wh[(i * 64 + h) * 64];
      for (int c = 0; c < 64; ++c) s += src[m * 64 + c] * wr[c];
      dst[e] = eluf_(s);
    }
    __syncthreads();
    float* tp = src; src = dst; dst = tp;
  }
  for (int e = tid; e < 4096; e += 256) {          // h0 (unmasked)
    int m = e >> 8, u = e & 255;
    float s = pe_bh0[u];
    for (int c = 0; c < 64; ++c) s += src[m * 64 + c] * pe_wh0[u * 64 + c];
    h32[m * 256 + u] = s;
    hA[((u >> 5) << 9) + a_addr(m, u & 31)] = (_Float16)s;
  }
  if (tid < 80) {                                   // z0
    int m = tid / 5, zd = tid % 5;
    float s = pe_bz0[zd];
    for (int c = 0; c < 64; ++c) s += src[m * 64 + c] * pe_wz0[zd * 64 + c];
    zbuf[tid] = s;
  }
  for (int e = tid; e < 512; e += 256) {            // consts section of inA
    int m = e >> 5, c = e & 31;
    int K = 265 + c;
    inA[((K >> 5) << 9) + a_addr(m, K & 31)] = (_Float16)cons32[m * 32 + c];
  }
  for (int e = tid; e < 16 * 23; e += 256) {        // zero pad K 297..319
    int m = e / 23, k = e % 23;
    int K = 297 + k;
    inA[((K >> 5) << 9) + a_addr(m, K & 31)] = (_Float16)0.f;
  }
  __syncthreads();

  int woff = 0;  // always 0; laundered per phase to defeat LICM of weight loads

  // ---- sequential time loop ----
  for (int t = 0; t < T_; ++t) {
    for (int e = tid; e < 1024; e += 256) {
      int m = e >> 6, f = e & 63;
      float v = traj[((size_t)(b0 + m) * T_ + t) * FULL_ + f];
      xA[((f >> 5) << 9) + a_addr(m, f & 31)] = (_Float16)v;
    }
    if (tid < 144) {
      int m = tid / 9, k = tid % 9;
      float v = (k < 5) ? zbuf[m * 5 + k] : aprev[m * 4 + (k - 5)];
      inA[a_addr(m, k)] = (_Float16)v;
    }
    __syncthreads();

    // ---- GRU step: wave wv owns hidden columns [wv*32, wv*32+32) ----
    float hnew[2][8];
#pragma clang loop unroll(disable)
    for (int nt2 = 0; nt2 < 2; ++nt2) {
      LAUNDER(woff);
      const v16h* Wt = Wv + woff;
      int nt = wv * 2 + nt2;
      int u  = nt * 16 + (lane & 15);
      float br  = gru_bih[u]       + gru_bhh[u];
      float bz  = gru_bih[256 + u] + gru_bhh[256 + u];
      float bni = gru_bih[512 + u];
      float bnh = gru_bhh[512 + u];
      v8f aR = {}, aZ = {}, aNi = {}, aNh = {};
#pragma unroll
      for (int j = 0; j < 8; ++j) { aR[j] = br; aZ[j] = bz; aNi[j] = bni; aNh[j] = bnh; }
#pragma unroll
      for (int kt = 0; kt < 2; ++kt) {                 // x contribution
        v16h a = *(const v16h*)&xA[(kt << 9) + (lane << 4)];
        aR  = wmma16(a, Wt[(kt * 16 + nt) * 32 + lane], aR);
        aZ  = wmma16(a, Wt[(160 + kt * 16 + nt) * 32 + lane], aZ);
        aNi = wmma16(a, Wt[(320 + kt * 16 + nt) * 32 + lane], aNi);
      }
#pragma unroll
      for (int kt = 0; kt < 8; ++kt) {                 // h contribution
        v16h a = *(const v16h*)&hA[(kt << 9) + (lane << 4)];
        aR  = wmma16(a, Wt[((kt + 2) * 16 + nt) * 32 + lane], aR);
        aZ  = wmma16(a, Wt[(160 + (kt + 2) * 16 + nt) * 32 + lane], aZ);
        aNh = wmma16(a, Wt[(352 + kt * 16 + nt) * 32 + lane], aNh);
      }
#pragma unroll
      for (int j = 0; j < 8; ++j) {
        int m = j + ((lane >> 4) << 3);
        float r  = sigmoidf_(aR[j]);
        float z  = sigmoidf_(aZ[j]);
        float n  = tanhf(aNi[j] + r * aNh[j]);
        float ho = h32[m * 256 + u];
        hnew[nt2][j] = (1.f - z) * n + z * ho;
      }
    }
    __syncthreads();  // all hA/xA reads complete before updates
#pragma clang loop unroll(disable)
    for (int nt2 = 0; nt2 < 2; ++nt2) {
      int nt = wv * 2 + nt2;
      int u  = nt * 16 + (lane & 15);
#pragma unroll
      for (int j = 0; j < 8; ++j) {
        int m = j + ((lane >> 4) << 3);
        float v = hnew[nt2][j];
        h32[m * 256 + u] = v;
        hA[((u >> 5) << 9) + a_addr(m, u & 31)] = (_Float16)v;
        int K = 9 + u;
        float vm = (t < slen[m]) ? v : 0.f;     // masked h feeds transition
        inA[((K >> 5) << 9) + a_addr(m, K & 31)] = (_Float16)vm;
      }
    }
    __syncthreads();

    // ---- transition MLP: wave wv owns HID columns [wv*16, wv*16+16) ----
    {
      LAUNDER(woff);
      const v16h* Wt = Wv + woff;
      int col = wv * 16 + (lane & 15);
      float bi = t_bin[col];
      v8f acc = {};
#pragma unroll
      for (int j = 0; j < 8; ++j) acc[j] = bi;
#pragma unroll
      for (int kt = 0; kt < 10; ++kt) {
        v16h a = *(const v16h*)&inA[(kt << 9) + (lane << 4)];
        acc = wmma16(a, Wt[(480 + kt * 8 + wv) * 32 + lane], acc);
      }
#pragma unroll
      for (int j = 0; j < 8; ++j) {                    // input layer: linear
        int m = j + ((lane >> 4) << 3);
        actA[((col >> 5) << 9) + a_addr(m, col & 31)] = (_Float16)acc[j];
      }
    }
    __syncthreads();
    {
      _Float16* cur = actA;
      _Float16* nxt = actB;
#pragma clang loop unroll(disable)
      for (int L = 0; L < NL_; ++L) {
        LAUNDER(woff);
        const v16h* Wt = Wv + woff;
        int col = wv * 16 + (lane & 15);
        float bi = t_bh[L * 128 + col];
        v8f acc = {};
#pragma unroll
        for (int j = 0; j < 8; ++j) acc[j] = bi;
#pragma unroll
        for (int kt = 0; kt < 4; ++kt) {
          v16h a = *(const v16h*)&cur[(kt << 9) + (lane << 4)];
          acc = wmma16(a, Wt[(560 + L * 32 + kt * 8 + wv) * 32 + lane], acc);
        }
#pragma unroll
        for (int j = 0; j < 8; ++j) {
          int m = j + ((lane >> 4) << 3);
          float v = eluf_(acc[j]);
          nxt[((col >> 5) << 9) + a_addr(m, col & 31)] = (_Float16)v;
          if (L == NL_ - 1) act32[m * 128 + col] = v;
        }
        __syncthreads();
        _Float16* tp = cur; cur = nxt; nxt = tp;
      }
    }

    // ---- z head (ZD=5 outputs) ----
    if (tid < 80) {
      int m = tid / 5, zd = tid % 5;
      float s = t_bz[zd];
      const float* wr = &t_wz[zd * 128];
      for (int k2 = 0; k2 < 128; ++k2) s += act32[m * 128 + k2] * wr[k2];
      zbuf[tid] = s;  // becomes z_prev for t+1
    }
    __syncthreads();

    // ---- mechanistic observation model + a_prev update ----
    if (tid < 16) {
      int m = tid;
      float hr  = zbuf[m * 5 + 0] + 70.f;
      float ra  = zbuf[m * 5 + 1] + 1.f;
      float ca  = zbuf[m * 5 + 2] + 2.f;
      float ttv = zbuf[m * 5 + 3] + 0.7f;
      float sv  = zbuf[m * 5 + 4] + 70.f;
      float e   = __expf(-ttv / (ra * ca));
      float inv = 1.f / (1.f - e);
      size_t o = ((size_t)(b0 + m) * T_ + t) * 4;
      out[o + 0] = hr;
      out[o + 1] = sv / ca * inv;
      out[o + 2] = e * (sv / ca) * inv;
      out[o + 3] = sv / 60.f * hr * ra;
    }
    if (tid < 64) {
      int m = tid >> 2, j = tid & 3;
      aprev[tid] = treat[((size_t)(b0 + m) * T_ + t) * 4 + j];
    }
    __syncthreads();
  }
}

// ---------------------------------------------------------------------------
extern "C" void kernel_launch(void* const* d_in, const int* in_sizes, int n_in,
                              void* d_out, int out_size, void* d_ws, size_t ws_size,
                              hipStream_t stream) {
  (void)in_sizes; (void)n_in; (void)out_size; (void)ws_size;
  const float* traj    = (const float*)d_in[0];
  const float* treat   = (const float*)d_in[1];
  const float* consts  = (const float*)d_in[2];
  const int*   slens   = (const int*)d_in[4];
  // params flattened in sorted-key (jax pytree) order:
  const float* gru_bhh = (const float*)d_in[6];
  const float* gru_bih = (const float*)d_in[7];
  const float* gru_whh = (const float*)d_in[8];
  const float* gru_wih = (const float*)d_in[9];
  const float* pe_b0   = (const float*)d_in[10];
  const float* pe_bh   = (const float*)d_in[11];
  const float* pe_bh0  = (const float*)d_in[12];
  const float* pe_bz0  = (const float*)d_in[13];
  const float* pe_w0   = (const float*)d_in[14];
  const float* pe_wh   = (const float*)d_in[15];
  const float* pe_wh0  = (const float*)d_in[16];
  const float* pe_wz0  = (const float*)d_in[17];
  const float* t_bh    = (const float*)d_in[18];
  const float* t_bin   = (const float*)d_in[19];
  const float* t_bz    = (const float*)d_in[20];
  const float* t_wh    = (const float*)d_in[21];
  const float* t_win   = (const float*)d_in[22];
  const float* t_wz    = (const float*)d_in[23];

  _Float16* W = (_Float16*)d_ws;  // 816*512 halves = 835584 bytes
  cvt_weights<<<(816 * 512 + 255) / 256, 256, 0, stream>>>(gru_whh, gru_wih,
                                                           t_wh, t_win, W);
  ssm_main<<<64, 256, 0, stream>>>(traj, treat, consts, slens,
                                   gru_bih, gru_bhh,
                                   pe_w0, pe_b0, pe_wh, pe_bh,
                                   pe_wh0, pe_bh0, pe_wz0, pe_bz0,
                                   t_bin, t_bh, t_wz, t_bz,
                                   W, (float*)d_out);
}
